// MinimalCrossObjectEncoder_5153960755948
// MI455X (gfx1250) — matrix operand order repaired and checked by reading, not compiled, not run on hardware
//
#include <hip/hip_runtime.h>
#include <hip/hip_bf16.h>

// ---------------- problem constants ----------------
constexpr int kB   = 16;    // batches
constexpr int kN   = 512;   // nodes per batch
constexpr int kIN  = 256;
constexpr int kHID = 128;
constexpr int kOUT = 256;
constexpr int kK   = 16;    // neighbors
constexpr int kM   = kB * kN;       // 8192 total rows
constexpr float kEPS = 1e-5f;

typedef __bf16    v16bf __attribute__((ext_vector_type(16)));
typedef float     v8f   __attribute__((ext_vector_type(8)));
typedef uint32_t  u32x4 __attribute__((ext_vector_type(4)));
typedef uint32_t  u32x8 __attribute__((ext_vector_type(8)));

// ---------------- WMMA helpers (gfx1250, wave32) ----------------
__device__ __forceinline__ v8f wmma_bf16(v16bf a, v16bf b, v8f c) {
    return __builtin_amdgcn_wmma_f32_16x16x32_bf16(
        false, a, false, b, (short)0, c, false, false);
}

// A-matrix 16x32 bf16 (ISA 7.12.2): lane m=lane&15, half=lane>>4,
// element e -> K = ((e&8)<<1) + half*8 + (e&7)  (K-contiguous in pairs of 8 -> b128 loads)
__device__ __forceinline__ v16bf load_A_f32(const float* src, int ld) {
    int lane = threadIdx.x & 31;
    int m = lane & 15, half = lane >> 4;
    v16bf a;
#pragma unroll
    for (int e = 0; e < 16; ++e) {
        int k = ((e & 8) << 1) + (half << 3) + (e & 7);
        a[e] = (__bf16)src[m * ld + k];
    }
    return a;
}
__device__ __forceinline__ v16bf load_A_bf(const __bf16* src, int ld) {
    int lane = threadIdx.x & 31;
    int m = lane & 15, half = lane >> 4;
    v16bf a;
#pragma unroll
    for (int e = 0; e < 16; ++e) {
        int k = ((e & 8) << 1) + (half << 3) + (e & 7);
        a[e] = src[m * ld + k];
    }
    return a;
}
// B-matrix 32x16, sourced from an N-major ("transposed") matrix: element e -> K = ((e&8)<<1) + half*8 + (e&7),
// B[k][n] = src[n*ld + k]  -> each lane reads 2 x 16 contiguous bytes (2 x b128).
__device__ __forceinline__ v16bf load_B_col_bf(const __bf16* src, size_t ld) {
    int lane = threadIdx.x & 31;
    int n = lane & 15, half = lane >> 4;
    v16bf b;
#pragma unroll
    for (int e = 0; e < 16; ++e) {
        int k = ((e & 8) << 1) + (half << 3) + (e & 7);
        b[e] = src[(size_t)n * ld + k];
    }
    return b;
}

__device__ __forceinline__ uint32_t pk2(float a, float b) {
    union { __bf16 h[2]; uint32_t u; } u;
    u.h[0] = (__bf16)a; u.h[1] = (__bf16)b;
    return u.u;
}

// ---------------- kernel 0: pack weights (transpose + bf16) ----------------
__global__ void pack_weights_kernel(const float* __restrict__ Wq, const float* __restrict__ Wk,
                                    const float* __restrict__ Wv, const float* __restrict__ Wr,
                                    const float* __restrict__ Wc,
                                    __bf16* __restrict__ WqT, __bf16* __restrict__ WkT,
                                    __bf16* __restrict__ WvT, __bf16* __restrict__ WrT,
                                    __bf16* __restrict__ WtT, __bf16* __restrict__ WbT)
{
    const int idx = blockIdx.x * 256 + threadIdx.x;     // 0..32767
    {   // [HID][IN] <- [IN][HID]
        const int h = idx >> 8, i = idx & 255;
        WqT[h * kIN + i] = (__bf16)Wq[i * kHID + h];
        WkT[h * kIN + i] = (__bf16)Wk[i * kHID + h];
        WvT[h * kIN + i] = (__bf16)Wv[i * kHID + h];
        WrT[h * kIN + i] = (__bf16)Wr[i * kHID + h];
    }
    {   // [OUT][HID] <- Wc split
        const int o = idx >> 7, k = idx & 127;
        const float top = Wc[k * kOUT + o];
        const float bot = Wc[(kHID + k) * kOUT + o];
        WtT[o * kHID + k] = (__bf16)(top - bot);
        WbT[o * kHID + k] = (__bf16)bot;
    }
}

// ---------------- kernel 1: projections; x staged via TDM ----------------
__global__ void proj_kernel(const float* __restrict__ x,
                            const __bf16* __restrict__ WqT, const __bf16* __restrict__ WkT,
                            const __bf16* __restrict__ WvT, const __bf16* __restrict__ WrT,
                            const float* __restrict__ bq, const float* __restrict__ bk,
                            const float* __restrict__ bv, const float* __restrict__ br,
                            __bf16* __restrict__ Qb, __bf16* __restrict__ Kb,
                            __bf16* __restrict__ VT, float* __restrict__ RT)
{
    __shared__ float xs[16 * kIN];                 // 16 KB x-tile (TDM destination)
    __shared__ float outs[16 * kHID];              // 8 KB C-layout staging
    const int row0 = blockIdx.x * 16;
    const int tid  = threadIdx.x;

    // --- Tensor Data Mover: DMA the 16x256 f32 tile x[row0:row0+16, :] into LDS ---
    if (tid < 32) {   // one wave issues; TDM ignores EXEC, tracked by TENSORcnt
        const uint64_t ga  = (uint64_t)(uintptr_t)(x + (size_t)row0 * kIN);
        const uint32_t lds = (uint32_t)(uintptr_t)xs;
        u32x4 g0;
        g0[0] = 1u;                                             // count=1 (user descriptor)
        g0[1] = lds;                                            // lds_addr
        g0[2] = (uint32_t)ga;                                   // global_addr[31:0]
        g0[3] = (uint32_t)((ga >> 32) & 0x01FFFFFFu) | (2u << 30); // addr[56:32], type=2
        u32x8 g1;
        g1[0] = (2u << 16);                                     // data_size = 4 bytes
        g1[1] = ((uint32_t)kIN) << 16;                          // tensor_dim0 = 256
        g1[2] = ((uint32_t)kM) << 16;                           // tensor_dim1 = 8192
        g1[3] = ((uint32_t)kIN) << 16;                          // tile_dim0 = 256
        g1[4] = 16u;                                            // tile_dim1 = 16
        g1[5] = (uint32_t)kIN;                                  // tensor_dim0_stride = 256
        g1[6] = 0u;
        g1[7] = 0u;
        asm volatile("tensor_load_to_lds %0, %1" :: "s"(g0), "s"(g1) : "memory");
        asm volatile("s_wait_tensorcnt 0x0" ::: "memory");
    }
    __syncthreads();

    const int wave = tid >> 5, lane = tid & 31;
    const int n = lane & 15, half = lane >> 4;
    const int col0 = wave * 16;                    // 8 waves cover HID=128

    const __bf16* Ws[4] = {WqT, WkT, WvT, WrT};
    const float*  bs[4] = {bq, bk, bv, br};
#pragma unroll
    for (int mtx = 0; mtx < 4; ++mtx) {
        v8f acc = {};
#pragma unroll
        for (int kk = 0; kk < kIN / 32; ++kk) {
            v16bf a  = load_A_f32(xs + kk * 32, kIN);
            v16bf bm = load_B_col_bf(Ws[mtx] + (size_t)col0 * kIN + kk * 32, kIN);
            acc = wmma_bf16(a, bm, acc);
        }
        const float bias = bs[mtx][col0 + n];
        __syncthreads();                           // previous cooperative readers done
#pragma unroll
        for (int i = 0; i < 8; ++i)
            outs[(i + half * 8) * kHID + col0 + n] = acc[i] + bias;
        __syncthreads();

        if (mtx <= 1) {                            // Q, K: row-major bf16, b128 stores
            __bf16* dst = (mtx == 0) ? Qb : Kb;
            const int r = tid >> 4, c0 = (tid & 15) * 8;
            const float* s = &outs[r * kHID + c0];
            uint4 w4;
            w4.x = pk2(s[0], s[1]); w4.y = pk2(s[2], s[3]);
            w4.z = pk2(s[4], s[5]); w4.w = pk2(s[6], s[7]);
            *(uint4*)(dst + (size_t)(row0 + r) * kHID + c0) = w4;
        } else if (mtx == 2) {                     // V transposed: VT[hid][token]
            const int c = tid >> 1, r0 = (tid & 1) * 8;
            uint4 w4;
            w4.x = pk2(outs[(r0 + 0) * kHID + c], outs[(r0 + 1) * kHID + c]);
            w4.y = pk2(outs[(r0 + 2) * kHID + c], outs[(r0 + 3) * kHID + c]);
            w4.z = pk2(outs[(r0 + 4) * kHID + c], outs[(r0 + 5) * kHID + c]);
            w4.w = pk2(outs[(r0 + 6) * kHID + c], outs[(r0 + 7) * kHID + c]);
            *(uint4*)(VT + (size_t)c * kM + row0 + r0) = w4;
        } else {                                   // R transposed f32: RT[hid][token]
            const int c = tid >> 1, r0 = (tid & 1) * 8;
            float4 a0, a1;
            a0.x = outs[(r0 + 0) * kHID + c]; a0.y = outs[(r0 + 1) * kHID + c];
            a0.z = outs[(r0 + 2) * kHID + c]; a0.w = outs[(r0 + 3) * kHID + c];
            a1.x = outs[(r0 + 4) * kHID + c]; a1.y = outs[(r0 + 5) * kHID + c];
            a1.z = outs[(r0 + 6) * kHID + c]; a1.w = outs[(r0 + 7) * kHID + c];
            *(float4*)(RT + (size_t)c * kM + row0 + r0)     = a0;
            *(float4*)(RT + (size_t)c * kM + row0 + r0 + 4) = a1;
        }
    }
}

// ---------------- kernel 2: fused attention ----------------
__global__ void attn_kernel(const __bf16* __restrict__ Qb, const __bf16* __restrict__ Kb,
                            const __bf16* __restrict__ VT, const float* __restrict__ RT,
                            __bf16* __restrict__ Hb, float* __restrict__ sqv)
{
    __shared__ float  S[16 * kN];                  // 32 KB scores (later reused as h staging)
    __shared__ __bf16 P[16 * kN];                  // 16 KB softmax weights
    __shared__ float  red[16 * 16];
    __shared__ float  sqacc[16];

    const int blk = blockIdx.x;
    const int b = blk >> 5, it = blk & 31;
    const int row0 = b * kN + it * 16;
    const int tid = threadIdx.x, wave = tid >> 5, lane = tid & 31;
    const int n = lane & 15, half = lane >> 4;

    // phase 1: S = (Q @ K^T) / sqrt(HID); A tiles hoisted out of the column loop
    v16bf aq[4];
#pragma unroll
    for (int kk = 0; kk < kHID / 32; ++kk)
        aq[kk] = load_A_bf(Qb + (size_t)row0 * kHID + kk * 32, kHID);

    for (int jt = wave; jt < kN / 16; jt += 8) {
        const int j0 = jt * 16;
        v8f acc = {};
#pragma unroll
        for (int kk = 0; kk < kHID / 32; ++kk) {
            v16bf bm = load_B_col_bf(Kb + (size_t)(b * kN + j0) * kHID + kk * 32, kHID);
            acc = wmma_bf16(aq[kk], bm, acc);
        }
        const float scale = 0.08838834764831845f;  // 1/sqrt(128)
#pragma unroll
        for (int i = 0; i < 8; ++i)
            S[(i + half * 8) * kN + j0 + n] = acc[i] * scale;
    }
    __syncthreads();

    // phase 2: row softmax (16 threads per row)
    const int r = tid >> 4, c = tid & 15;
    float mx = -3.4e38f;
    for (int j = c; j < kN; j += 16) mx = fmaxf(mx, S[r * kN + j]);
    red[r * 16 + c] = mx;
    __syncthreads();
    if (c == 0) {
        float m = red[r * 16];
        for (int t = 1; t < 16; ++t) m = fmaxf(m, red[r * 16 + t]);
        red[r * 16] = m;
    }
    __syncthreads();
    mx = red[r * 16];
    float sum = 0.f;
    for (int j = c; j < kN; j += 16) {
        float e = __expf(S[r * kN + j] - mx);
        S[r * kN + j] = e;
        sum += e;
    }
    __syncthreads();
    red[r * 16 + c] = sum;
    __syncthreads();
    if (c == 0) {
        float s = 0.f;
        for (int t = 0; t < 16; ++t) s += red[r * 16 + t];
        red[r * 16] = s;
    }
    __syncthreads();
    const float inv = 1.f / red[r * 16];
    for (int j = c; j < kN; j += 16)
        P[r * kN + j] = (__bf16)(S[r * kN + j] * inv);
    if (tid < 16) sqacc[tid] = 0.f;
    __syncthreads();

    // phase 3: h = P @ V + R  (B operand from transposed VT -> contiguous b128 loads)
    const int col0 = wave * 16;
    v8f acc3 = {};
#pragma unroll
    for (int kt = 0; kt < kN / 32; ++kt) {
        v16bf a  = load_A_bf(P + kt * 32, kN);
        v16bf bm = load_B_col_bf(VT + (size_t)col0 * kM + (b * kN + kt * 32), kM);
        acc3 = wmma_bf16(a, bm, acc3);
    }
    float* Hs = S;                                 // reuse score buffer as f32 h staging
#pragma unroll
    for (int i = 0; i < 8; ++i) {
        const int m = i + half * 8;
        Hs[m * kHID + col0 + n] = acc3[i] + RT[(size_t)(col0 + n) * kM + row0 + m];
    }
    __syncthreads();
    {   // cooperative: Hb row-major b128 stores + ||h||^2 per row
        const int rr = tid >> 4, c0 = (tid & 15) * 8;
        const float* s = &Hs[rr * kHID + c0];
        float part = 0.f;
#pragma unroll
        for (int i = 0; i < 8; ++i) part += s[i] * s[i];
        atomicAdd(&sqacc[rr], part);
        uint4 w4;
        w4.x = pk2(s[0], s[1]); w4.y = pk2(s[2], s[3]);
        w4.z = pk2(s[4], s[5]); w4.w = pk2(s[6], s[7]);
        *(uint4*)(Hb + (size_t)(row0 + rr) * kHID + c0) = w4;
    }
    __syncthreads();
    if (tid < 16) sqv[row0 + tid] = sqacc[tid];
}

// ---------------- kernel 3: fused Gram tile + top-16 nearest ----------------
__global__ void dist_topk_kernel(const __bf16* __restrict__ Hb,
                                 const float* __restrict__ sqv,
                                 int* __restrict__ idx)
{
    __shared__ float G[16 * kN];                   // 32 KB Gram tile-row
    __shared__ float sqj[kN];

    const int blk = blockIdx.x;
    const int b = blk >> 5, it = blk & 31;
    const int row0 = b * kN + it * 16;
    const int tid = threadIdx.x, wave = tid >> 5, lane = tid & 31;
    const int n = lane & 15, half = lane >> 4;

    for (int j = tid; j < kN; j += 256) sqj[j] = sqv[b * kN + j];

    v16bf ah[4];
#pragma unroll
    for (int kk = 0; kk < kHID / 32; ++kk)
        ah[kk] = load_A_bf(Hb + (size_t)row0 * kHID + kk * 32, kHID);

    for (int jt = wave; jt < kN / 16; jt += 8) {
        const int j0 = jt * 16;
        v8f acc = {};
#pragma unroll
        for (int kk = 0; kk < kHID / 32; ++kk) {
            v16bf bm = load_B_col_bf(Hb + (size_t)(b * kN + j0) * kHID + kk * 32, kHID);
            acc = wmma_bf16(ah[kk], bm, acc);
        }
#pragma unroll
        for (int i = 0; i < 8; ++i)
            G[(i + half * 8) * kN + j0 + n] = acc[i];
    }
    __syncthreads();

    if (tid < 16) {
        const int rr = tid;
        const float si = sqv[row0 + rr];
        float bd[kK]; int bi[kK];
#pragma unroll
        for (int t = 0; t < kK; ++t) { bd[t] = 3.4e38f; bi[t] = 0; }
        for (int j = 0; j < kN; ++j) {
            const float d = si + sqj[j] - 2.f * G[rr * kN + j];
            if (d < bd[kK - 1]) {                  // stable: earlier index wins ties
                int t = kK - 1;
                while (t > 0 && bd[t - 1] > d) {
                    bd[t] = bd[t - 1]; bi[t] = bi[t - 1]; --t;
                }
                bd[t] = d; bi[t] = j;
            }
        }
        for (int t = 0; t < kK; ++t) idx[(size_t)(row0 + rr) * kK + t] = bi[t];
    }
}

// ---------------- kernel 4: base = h@(Wc_top-Wc_bot)+bc ; p = h@Wc_bot ----------------
__global__ void basep_kernel(const __bf16* __restrict__ Hb,
                             const __bf16* __restrict__ WtT, const __bf16* __restrict__ WbT,
                             const float* __restrict__ bc,
                             float* __restrict__ basev, float* __restrict__ pv)
{
    __shared__ float bo[16 * kOUT];                // 16 KB
    __shared__ float po[16 * kOUT];                // 16 KB
    const int row0 = blockIdx.x * 16;
    const int tid = threadIdx.x, wave = tid >> 5, lane = tid & 31;
    const int n = lane & 15, half = lane >> 4;

    v16bf ah[4];
#pragma unroll
    for (int kk = 0; kk < kHID / 32; ++kk)
        ah[kk] = load_A_bf(Hb + (size_t)row0 * kHID + kk * 32, kHID);

    for (int nt = wave; nt < kOUT / 16; nt += 8) { // 2 output tiles per wave
        const int col0 = nt * 16;
        v8f accB = {}, accP = {};
#pragma unroll
        for (int kk = 0; kk < kHID / 32; ++kk) {
            v16bf bt = load_B_col_bf(WtT + (size_t)col0 * kHID + kk * 32, kHID);
            v16bf bb = load_B_col_bf(WbT + (size_t)col0 * kHID + kk * 32, kHID);
            accB = wmma_bf16(ah[kk], bt, accB);
            accP = wmma_bf16(ah[kk], bb, accP);
        }
        const float bias = bc[col0 + n];
#pragma unroll
        for (int i = 0; i < 8; ++i) {
            const int m = i + half * 8;
            bo[m * kOUT + col0 + n] = accB[i] + bias;
            po[m * kOUT + col0 + n] = accP[i];
        }
    }
    __syncthreads();
    {   // cooperative b128 stores
        const int r = tid >> 4, c0 = (tid & 15) * 16;
#pragma unroll
        for (int q = 0; q < 16; q += 4) {
            float4 v4 = *(const float4*)&bo[r * kOUT + c0 + q];
            *(float4*)(basev + (size_t)(row0 + r) * kOUT + c0 + q) = v4;
            float4 p4 = *(const float4*)&po[r * kOUT + c0 + q];
            *(float4*)(pv + (size_t)(row0 + r) * kOUT + c0 + q) = p4;
        }
    }
}

// ---------------- kernel 5: gather-max, LayerNorm, SELU ----------------
__global__ void final_kernel(const float* __restrict__ basev, const float* __restrict__ pv,
                             const int* __restrict__ idx,
                             const float* __restrict__ ln_scale, const float* __restrict__ ln_bias,
                             float* __restrict__ out)
{
    __shared__ float red[kOUT];
    const int row = blockIdx.x;
    const int b = row >> 9;
    const int o = threadIdx.x;

    float v = basev[(size_t)row * kOUT + o];
    const int* id = idx + (size_t)row * kK;
    float mk = -3.4e38f;
#pragma unroll
    for (int t = 0; t < kK; ++t) {
        const int j = id[t];
        mk = fmaxf(mk, pv[(size_t)(b * kN + j) * kOUT + o]);
    }
    v += mk;

    red[o] = v;
    __syncthreads();
    for (int s = kOUT / 2; s > 0; s >>= 1) { if (o < s) red[o] += red[o + s]; __syncthreads(); }
    const float mu = red[0] * (1.f / kOUT);
    __syncthreads();
    const float d = v - mu;
    red[o] = d * d;
    __syncthreads();
    for (int s = kOUT / 2; s > 0; s >>= 1) { if (o < s) red[o] += red[o + s]; __syncthreads(); }
    const float var = red[0] * (1.f / kOUT);

    float yn = d * rsqrtf(var + kEPS) * ln_scale[o] + ln_bias[o];
    const float sc = 1.0507009873554805f, al = 1.6732632423543772f;
    out[(size_t)row * kOUT + o] = yn > 0.f ? sc * yn : sc * al * (__expf(yn) - 1.f);
}

// ---------------- launcher ----------------
extern "C" void kernel_launch(void* const* d_in, const int* in_sizes, int n_in,
                              void* d_out, int out_size, void* d_ws, size_t ws_size,
                              hipStream_t stream) {
    const float* x  = (const float*)d_in[0];
    // d_in[1] = n_nodes (always N, unused)
    const float* Wq = (const float*)d_in[2];  const float* bq = (const float*)d_in[3];
    const float* Wk = (const float*)d_in[4];  const float* bk = (const float*)d_in[5];
    const float* Wv = (const float*)d_in[6];  const float* bv = (const float*)d_in[7];
    const float* Wr = (const float*)d_in[8];  const float* br = (const float*)d_in[9];
    const float* Wc = (const float*)d_in[10]; const float* bc = (const float*)d_in[11];
    const float* ls = (const float*)d_in[12]; const float* lb = (const float*)d_in[13];
    float* out = (float*)d_out;

    char* w = (char*)d_ws;
    __bf16* Qb  = (__bf16*)w; w += (size_t)kM * kHID * sizeof(__bf16);
    __bf16* Kb  = (__bf16*)w; w += (size_t)kM * kHID * sizeof(__bf16);
    __bf16* VT  = (__bf16*)w; w += (size_t)kM * kHID * sizeof(__bf16);
    __bf16* Hb  = (__bf16*)w; w += (size_t)kM * kHID * sizeof(__bf16);
    float*  RT  = (float*)w;  w += (size_t)kM * kHID * sizeof(float);
    float*  sq  = (float*)w;  w += (size_t)kM * sizeof(float);
    int*    ix  = (int*)w;    w += (size_t)kM * kK * sizeof(int);
    float*  bvv = (float*)w;  w += (size_t)kM * kOUT * sizeof(float);
    float*  pvv = (float*)w;  w += (size_t)kM * kOUT * sizeof(float);
    __bf16* WqT = (__bf16*)w; w += (size_t)kHID * kIN * sizeof(__bf16);
    __bf16* WkT = (__bf16*)w; w += (size_t)kHID * kIN * sizeof(__bf16);
    __bf16* WvT = (__bf16*)w; w += (size_t)kHID * kIN * sizeof(__bf16);
    __bf16* WrT = (__bf16*)w; w += (size_t)kHID * kIN * sizeof(__bf16);
    __bf16* WtT = (__bf16*)w; w += (size_t)kOUT * kHID * sizeof(__bf16);
    __bf16* WbT = (__bf16*)w; w += (size_t)kOUT * kHID * sizeof(__bf16);

    const int mtiles = kM / 16;                          // 512
    pack_weights_kernel<<<128, 256, 0, stream>>>(Wq, Wk, Wv, Wr, Wc,
                                                 WqT, WkT, WvT, WrT, WtT, WbT);
    proj_kernel<<<mtiles, 256, 0, stream>>>(x, WqT, WkT, WvT, WrT, bq, bk, bv, br,
                                            Qb, Kb, VT, RT);
    attn_kernel<<<kB * (kN / 16), 256, 0, stream>>>(Qb, Kb, VT, RT, Hb, sq);
    dist_topk_kernel<<<kB * (kN / 16), 256, 0, stream>>>(Hb, sq, ix);
    basep_kernel<<<mtiles, 256, 0, stream>>>(Hb, WtT, WbT, bc, bvv, pvv);
    final_kernel<<<kM, 256, 0, stream>>>(bvv, pvv, ix, ls, lb, out);
}
